// HardAttentionMemoryAE_39204461478055
// MI455X (gfx1250) — matrix operand order repaired and checked by reading, not compiled, not run on hardware
//
#include <hip/hip_runtime.h>
#include <hip/hip_bf16.h>

#if __has_include(<hip/amd_detail/amd_gfx1250_TDM.h>)
#define TDM_6ARG 1
#endif

// ---------------------------------------------------------------------------
// Types for CDNA5 WMMA / TDM
// ---------------------------------------------------------------------------
typedef __attribute__((ext_vector_type(16))) __bf16         v16bf;
typedef __attribute__((ext_vector_type(8)))  float          v8f;
typedef __attribute__((ext_vector_type(8)))  unsigned int   v8u;
typedef __attribute__((ext_vector_type(4)))  unsigned int   tdm_g0_t;
typedef __attribute__((ext_vector_type(8)))  int            tdm_g1_t;
typedef __attribute__((ext_vector_type(4)))  int            tdm_g2_t;

#define B_ROWS   65536
#define IN_DIM   784
#define HID_DIM  256
#define EMB_DIM  128
#define MEM_SZ   50

__device__ __forceinline__ unsigned short f2bf(float f) {
    unsigned int u = __float_as_uint(f);
    u += 0x7fffu + ((u >> 16) & 1u);     // round-to-nearest-even
    return (unsigned short)(u >> 16);
}

// ---------------------------------------------------------------------------
// TDM: async DMA of a 64(n) x 32(k) bf16 tile of row-major Wt[N][K] into LDS.
// x-dim (contiguous) is K.  LDS padding: 1 DWORD after every 16 DWORDs
// (pad_interval=3, pad_amount=0) -> LDS row stride = 34 u16, matching the
// bank-friendly [col][k] layout the WMMA B fragments read with b32 loads.
// OOB (ragged K/N edges) zero-filled by hardware (dims = remaining extent).
// ---------------------------------------------------------------------------
__device__ __forceinline__ void tdm_load_bt_tile(const unsigned short* Bt,
                                                 unsigned lds_off,
                                                 int k0, int n0, int N, int K) {
    unsigned long long ga =
        (unsigned long long)(const unsigned char*)(Bt + (size_t)n0 * K + k0);
    unsigned td0 = (unsigned)(K - k0);           // remaining extent along x (k)
    unsigned td1 = (unsigned)(N - n0);           // remaining extent along y (n)
    unsigned long long s0 = (unsigned long long)(unsigned)K;  // elem stride

    tdm_g0_t g0 = {
        1u,                                                    // count=1
        lds_off,                                               // lds_addr
        (unsigned)(ga & 0xffffffffu),                          // gaddr[31:0]
        (unsigned)((ga >> 32) & 0x01ffffffu) | (2u << 30)      // gaddr[56:32]|type=2
    };
    tdm_g1_t g1 = {
        (int)((1u << 16) | (1u << 20) | (3u << 22)),           // 2B, pad_en, ivl=16dw
        (int)((td0 & 0xffffu) << 16),                          // tdim0[15:0]
        (int)(((td0 >> 16) & 0xffffu) | ((td1 & 0xffffu) << 16)),
        (int)(((td1 >> 16) & 0xffffu) | (32u << 16)),          // tile_dim0=32 (k)
        (int)64,                                               // tile_dim1=64 (n)
        (int)(unsigned)(s0 & 0xffffffffu),                     // stride0[31:0]
        (int)(unsigned)((s0 >> 32) & 0xffffu),                 // stride0[47:32]
        0
    };
    tdm_g2_t gz = {0, 0, 0, 0};
#ifdef TDM_6ARG
    tdm_g1_t z8 = {0, 0, 0, 0, 0, 0, 0, 0};
    __builtin_amdgcn_tensor_load_to_lds(g0, g1, gz, gz, z8, 0);
#else
    __builtin_amdgcn_tensor_load_to_lds(g0, g1, gz, gz, 0);
#endif
}

// ---------------------------------------------------------------------------
// Prep kernels
// ---------------------------------------------------------------------------
// src[K][N] f32  ->  dst[N][K] bf16   (one-time transposed weight conversion)
__global__ void cvt_bf16_t_kernel(const float* __restrict__ s,
                                  unsigned short* __restrict__ d, int K, int N) {
    int i = blockIdx.x * 256 + threadIdx.x;
    if (i < K * N) {
        int k = i / N, n = i - k * N;
        d[(size_t)n * K + k] = f2bf(s[i]);
    }
}

__global__ void mem_norm_kernel(const float* __restrict__ memf,
                                float* __restrict__ mem_norm) {
    __shared__ float red[EMB_DIM];
    int m = blockIdx.x, k = threadIdx.x;
    float v = memf[m * EMB_DIM + k];
    red[k] = v * v;
    __syncthreads();
    for (int s = EMB_DIM / 2; s > 0; s >>= 1) {
        if (k < s) red[k] += red[k + s];
        __syncthreads();
    }
    float scale = 1.0f / fmaxf(sqrtf(red[0]), 1e-12f);
    mem_norm[m * EMB_DIM + k] = v * scale;
}

__global__ void mem_sum_kernel(const float* __restrict__ memf,
                               float* __restrict__ sum_mem) {
    int k = threadIdx.x;
    float s = 0.f;
    for (int m = 0; m < MEM_SZ; ++m) s += memf[m * EMB_DIM + k];
    sum_mem[k] = s;
}

// ---------------------------------------------------------------------------
// WMMA GEMM:  C[M,N] = epi(A[M,K] * W[K,N] + bias), W given transposed bf16.
//   block = 256 thr = 8 waves ; block tile 128(M) x 64(N) ; K-step 32
//   B tile arrives via TDM into padded [col][k] LDS (wave 0 issues,
//   s_wait_tensorcnt + barrier).  A tile staged cooperatively.
//   EPI: 0 none, 1 relu, 2 sigmoid.
// ---------------------------------------------------------------------------
#define APAD 34
#define BKPAD 34   // 16 DWORDs data + 1 DWORD TDM pad per 32-elem row

template <int EPI, bool A_F32, bool OUT_BF16>
__global__ __launch_bounds__(256, 2)
void gemm_wmma_kernel(const void* __restrict__ Aptr,
                      const unsigned short* __restrict__ Bt,   // [N][K] bf16
                      const float* __restrict__ bias,
                      void* __restrict__ Cptr,
                      int M, int N, int K) {
    __shared__ unsigned short ldsA[128 * APAD];    // [row][k]
    __shared__ unsigned short ldsBt[64 * BKPAD];   // [col][k] (TDM dest, padded)

    const int tid  = threadIdx.x;
    const int wid  = tid >> 5;
    const int lane = tid & 31;
    const int hf   = lane >> 4;      // 0: lanes 0-15, 1: lanes 16-31
    const int r16  = lane & 15;

    const int m0 = blockIdx.y * 128;
    const int n0 = blockIdx.x * 64;

    const float*          Af = (const float*)Aptr;
    const unsigned short* Ab = (const unsigned short*)Aptr;

    const unsigned ldsB_off = (unsigned)(unsigned long long)(const void*)&ldsBt[0];

    v8f acc[4];
#pragma unroll
    for (int j = 0; j < 4; ++j)
#pragma unroll
        for (int i = 0; i < 8; ++i) acc[j][i] = 0.0f;

    const int kfull  = K >> 5;
    const int ksteps = kfull + ((K & 31) ? 1 : 0);

    for (int ks = 0; ks < ksteps; ++ks) {
        const int k0 = ks * 32;

        // ---- B tile: one TDM descriptor per block-iteration (wave 0) ----
        if (wid == 0)
            tdm_load_bt_tile(Bt, ldsB_off, k0, n0, N, K);

        // ---- prefetch next A K-tile (global_prefetch_b8) ----
        if (k0 + 32 < K) {
            int prow = tid & 127;
            size_t poff = (size_t)(m0 + prow) * K + (size_t)(k0 + 32) + ((tid >> 7) ? 16 : 0);
            if constexpr (A_F32) __builtin_prefetch(Af + poff, 0, 3);
            else                 __builtin_prefetch(Ab + poff, 0, 3);
        }

        // ---- stage A tile: 128 x 32 -> bf16 LDS ----
        if (ks < kfull) {
            // fast path: no K guard
#pragma unroll
            for (int it = 0; it < 4; ++it) {
                int idx = it * 256 + tid;        // 0..1023 quads
                int ar  = idx >> 3;              // 0..127
                int ac  = (idx & 7) * 4;         // 0..28
                unsigned short h0, h1, h2, h3;
                if constexpr (A_F32) {
                    const float* src = Af + (size_t)(m0 + ar) * K + (k0 + ac);
                    float4 v = *(const float4*)src;
                    h0 = f2bf(v.x); h1 = f2bf(v.y); h2 = f2bf(v.z); h3 = f2bf(v.w);
                } else {
                    const unsigned short* src = Ab + (size_t)(m0 + ar) * K + (k0 + ac);
                    unsigned long long p = *(const unsigned long long*)src;
                    h0 = (unsigned short)(p);
                    h1 = (unsigned short)(p >> 16);
                    h2 = (unsigned short)(p >> 32);
                    h3 = (unsigned short)(p >> 48);
                }
                unsigned short* dst = &ldsA[ar * APAD + ac];
                dst[0] = h0; dst[1] = h1; dst[2] = h2; dst[3] = h3;
            }
        } else {
            // ragged K tail (only once, only when K % 32 != 0)
#pragma unroll
            for (int it = 0; it < 4; ++it) {
                int idx = it * 256 + tid;
                int ar  = idx >> 3;
                int ac  = (idx & 7) * 4;
                unsigned short h[4] = {0, 0, 0, 0};
#pragma unroll
                for (int t = 0; t < 4; ++t) {
                    int gk = k0 + ac + t;
                    if (gk < K) {
                        if constexpr (A_F32)
                            h[t] = f2bf(Af[(size_t)(m0 + ar) * K + gk]);
                        else
                            h[t] = Ab[(size_t)(m0 + ar) * K + gk];
                    }
                }
                unsigned short* dst = &ldsA[ar * APAD + ac];
                dst[0] = h[0]; dst[1] = h[1]; dst[2] = h[2]; dst[3] = h[3];
            }
        }

        // wave 0 must see TDM completion before signaling the barrier
        if (wid == 0) __builtin_amdgcn_s_wait_tensorcnt(0);
        __syncthreads();

        // ---- A fragment (16-bit A 16x32: lanes0-15 K=0..15,
        //      lanes16-31 K=16..31; VGPR i holds K pair) ----
        const unsigned short* arow = &ldsA[(wid * 16 + r16) * APAD];
        const int kb = hf * 16;
        v8u araw;
        araw[0] = *(const unsigned int*)&arow[kb + 0];
        araw[1] = *(const unsigned int*)&arow[kb + 2];
        araw[2] = *(const unsigned int*)&arow[kb + 4];
        araw[3] = *(const unsigned int*)&arow[kb + 6];
        araw[4] = *(const unsigned int*)&arow[kb + 8];
        araw[5] = *(const unsigned int*)&arow[kb + 10];
        araw[6] = *(const unsigned int*)&arow[kb + 12];
        araw[7] = *(const unsigned int*)&arow[kb + 14];
        v16bf afrag = __builtin_bit_cast(v16bf, araw);

        // ---- 4 B fragments (B 32x16: lanes0-15 K=0..15, lanes16-31 K=16..31;
        //      col = lane%16), contiguous b32 reads from padded [col][k] ----
        v16bf bfr[4];
#pragma unroll
        for (int j = 0; j < 4; ++j) {
            const unsigned short* bp = &ldsBt[(j * 16 + r16) * BKPAD + hf * 16];
            v8u braw;
#pragma unroll
            for (int i = 0; i < 8; ++i)
                braw[i] = *(const unsigned int*)&bp[2 * i];
            bfr[j] = __builtin_bit_cast(v16bf, braw);
        }

        // ---- 4 back-to-back WMMAs ----
#pragma unroll
        for (int j = 0; j < 4; ++j)
            acc[j] = __builtin_amdgcn_wmma_f32_16x16x32_bf16(
                false, afrag, false, bfr[j], (short)0, acc[j], false, false);

        __syncthreads();
    }

    // ---- epilogue: C/D layout -> row = i + 8*hf, col = lane%16 ----
#pragma unroll
    for (int j = 0; j < 4; ++j) {
        int col = n0 + j * 16 + r16;
        if (col >= N) continue;
        float bv = bias[col];
#pragma unroll
        for (int i = 0; i < 8; ++i) {
            int row = m0 + wid * 16 + i + hf * 8;
            float v = acc[j][i] + bv;
            if constexpr (EPI == 1) v = fmaxf(v, 0.0f);
            if constexpr (EPI == 2) v = 1.0f / (1.0f + __expf(-v));
            if constexpr (OUT_BF16)
                ((unsigned short*)Cptr)[(size_t)row * N + col] = f2bf(v);
            else
                ((float*)Cptr)[(size_t)row * N + col] = v;
        }
    }
}

// ---------------------------------------------------------------------------
// Per-row hard-attention:  z -> l2norm -> cosine sims vs mem_norm -> top5 ->
// softmax-with-zeros -> z_mem (bf16).   One wave32 per row, 8 rows/block.
//   z_mem = (sum_mem + sum_top5 (e^v - 1) * memory[i]) / (45 + sum_top5 e^v)
// ---------------------------------------------------------------------------
__global__ __launch_bounds__(256, 2)
void attn_kernel(const float* __restrict__ z,          // [B,128] fp32
                 const float* __restrict__ memf,       // [50,128] fp32
                 const float* __restrict__ mem_norm,   // [50,128] fp32
                 const float* __restrict__ sum_mem,    // [128] fp32
                 unsigned short* __restrict__ zmem) {  // [B,128] bf16
    __shared__ float s_mn[MEM_SZ * EMB_DIM];
    __shared__ float s_mem[MEM_SZ * EMB_DIM];
    __shared__ float s_sum[EMB_DIM];

    const int tid = threadIdx.x;
    for (int i = tid; i < MEM_SZ * EMB_DIM; i += 256) {
        s_mn[i]  = mem_norm[i];
        s_mem[i] = memf[i];
    }
    if (tid < EMB_DIM) s_sum[tid] = sum_mem[tid];
    __syncthreads();

    const int wid  = tid >> 5;
    const int lane = tid & 31;
    const int row  = blockIdx.x * 8 + wid;

    const float* zr = z + (size_t)row * EMB_DIM;
    float z0 = zr[lane], z1 = zr[lane + 32], z2 = zr[lane + 64], z3 = zr[lane + 96];

    float ss = z0 * z0 + z1 * z1 + z2 * z2 + z3 * z3;
#pragma unroll
    for (int off = 16; off > 0; off >>= 1) ss += __shfl_xor(ss, off, 32);
    float scale = 1.0f / fmaxf(sqrtf(ss), 1e-12f);
    z0 *= scale; z1 *= scale; z2 *= scale; z3 *= scale;

    // cosine sims, distributed: lane l holds sim[l] (v1) and sim[l+32] (v2)
    float v1 = -1e30f, v2 = -1e30f;
    for (int m = 0; m < MEM_SZ; ++m) {
        const float* mr = &s_mn[m * EMB_DIM];
        float p = z0 * mr[lane] + z1 * mr[lane + 32] +
                  z2 * mr[lane + 64] + z3 * mr[lane + 96];
#pragma unroll
        for (int off = 16; off > 0; off >>= 1) p += __shfl_xor(p, off, 32);
        if (m < 32) { if (lane == m)      v1 = p; }
        else        { if (lane == m - 32) v2 = p; }
    }

    // top-5 via 5 argmax rounds (deterministic: ties -> smaller index)
    float denom = (float)(MEM_SZ - 5);   // 45 entries of exp(0)
    int   sel[5];
    float coef[5];
#pragma unroll
    for (int t = 0; t < 5; ++t) {
        float bv; int bi;
        if (v1 >= v2) { bv = v1; bi = lane; } else { bv = v2; bi = lane + 32; }
#pragma unroll
        for (int off = 16; off > 0; off >>= 1) {
            float ov = __shfl_xor(bv, off, 32);
            int   oi = __shfl_xor(bi, off, 32);
            if (ov > bv || (ov == bv && oi < bi)) { bv = ov; bi = oi; }
        }
        sel[t] = bi;
        float e = __expf(bv);
        denom += e;
        coef[t] = e - 1.0f;
        if (bi < 32) { if (lane == bi)      v1 = -1e30f; }
        else         { if (lane == bi - 32) v2 = -1e30f; }
    }

    float inv = 1.0f / denom;
    float o0 = s_sum[lane],      o1 = s_sum[lane + 32],
          o2 = s_sum[lane + 64], o3 = s_sum[lane + 96];
#pragma unroll
    for (int t = 0; t < 5; ++t) {
        const float* mr = &s_mem[sel[t] * EMB_DIM];
        float c = coef[t];
        o0 += c * mr[lane];      o1 += c * mr[lane + 32];
        o2 += c * mr[lane + 64]; o3 += c * mr[lane + 96];
    }

    unsigned short* orow = zmem + (size_t)row * EMB_DIM;
    orow[lane]      = f2bf(o0 * inv);
    orow[lane + 32] = f2bf(o1 * inv);
    orow[lane + 64] = f2bf(o2 * inv);
    orow[lane + 96] = f2bf(o3 * inv);
}

// ---------------------------------------------------------------------------
// Launcher
// ---------------------------------------------------------------------------
extern "C" void kernel_launch(void* const* d_in, const int* in_sizes, int n_in,
                              void* d_out, int out_size, void* d_ws, size_t ws_size,
                              hipStream_t stream) {
    (void)in_sizes; (void)n_in; (void)out_size; (void)ws_size;

    const float* x    = (const float*)d_in[0];
    const float* W1   = (const float*)d_in[1];
    const float* b1   = (const float*)d_in[2];
    const float* W2   = (const float*)d_in[3];
    const float* b2   = (const float*)d_in[4];
    const float* memf = (const float*)d_in[5];
    const float* W3   = (const float*)d_in[6];
    const float* b3   = (const float*)d_in[7];
    const float* W4   = (const float*)d_in[8];
    const float* b4   = (const float*)d_in[9];
    float* out = (float*)d_out;

    char* ws = (char*)d_ws;
    // weight / stat region (first 1 MB) -- all weights stored TRANSPOSED [N][K]
    unsigned short* W1t = (unsigned short*)(ws + 0);
    unsigned short* W2t = (unsigned short*)(ws + 401408);
    unsigned short* W3t = (unsigned short*)(ws + 466944);
    unsigned short* W4t = (unsigned short*)(ws + 532480);
    float* mem_norm     = (float*)(ws + 933888);
    float* sum_mem      = (float*)(ws + 959488);
    // big buffers
    unsigned short* bufA = (unsigned short*)(ws + (1u << 20));            // h / d  bf16 [B,256]
    float*          bufB = (float*)(ws + (1u << 20) + 33554432u);         // z      f32  [B,128]
    unsigned short* bufC = (unsigned short*)(ws + (1u << 20) + 67108864u);// z_mem  bf16 [B,128]

    // --- prep: transposed weight conversion + memory stats ---
    cvt_bf16_t_kernel<<<(IN_DIM * HID_DIM + 255) / 256, 256, 0, stream>>>(W1, W1t, IN_DIM, HID_DIM);
    cvt_bf16_t_kernel<<<(HID_DIM * EMB_DIM + 255) / 256, 256, 0, stream>>>(W2, W2t, HID_DIM, EMB_DIM);
    cvt_bf16_t_kernel<<<(EMB_DIM * HID_DIM + 255) / 256, 256, 0, stream>>>(W3, W3t, EMB_DIM, HID_DIM);
    cvt_bf16_t_kernel<<<(HID_DIM * IN_DIM + 255) / 256, 256, 0, stream>>>(W4, W4t, HID_DIM, IN_DIM);
    mem_norm_kernel<<<MEM_SZ, EMB_DIM, 0, stream>>>(memf, mem_norm);
    mem_sum_kernel<<<1, EMB_DIM, 0, stream>>>(memf, sum_mem);

    // --- stage 1: h = relu(x @ W1 + b1) -> bufA (bf16) ---
    gemm_wmma_kernel<1, true, true><<<dim3(HID_DIM / 64, B_ROWS / 128), 256, 0, stream>>>(
        x, W1t, b1, bufA, B_ROWS, HID_DIM, IN_DIM);

    // --- stage 2: z = h @ W2 + b2 -> bufB (f32) ---
    gemm_wmma_kernel<0, false, false><<<dim3(EMB_DIM / 64, B_ROWS / 128), 256, 0, stream>>>(
        bufA, W2t, b2, bufB, B_ROWS, EMB_DIM, HID_DIM);

    // --- stage 3: hard attention -> bufC (bf16 z_mem) ---
    attn_kernel<<<B_ROWS / 8, 256, 0, stream>>>(bufB, memf, mem_norm, sum_mem, bufC);

    // --- stage 4: d = relu(z_mem @ W3 + b3) -> bufA (bf16, h is dead) ---
    gemm_wmma_kernel<1, false, true><<<dim3(HID_DIM / 64, B_ROWS / 128), 256, 0, stream>>>(
        bufC, W3t, b3, bufA, B_ROWS, HID_DIM, EMB_DIM);

    // --- stage 5: x_hat = sigmoid(d @ W4 + b4) -> d_out (f32), ragged N ---
    gemm_wmma_kernel<2, false, false><<<dim3((IN_DIM + 63) / 64, B_ROWS / 128), 256, 0, stream>>>(
        bufA, W4t, b4, out, B_ROWS, IN_DIM, HID_DIM);
}
